// GraphAttentionLayer_72834055405802
// MI455X (gfx1250) — matrix operand order repaired and checked by reading, not compiled, not run on hardware
//
#include <hip/hip_runtime.h>
#include <hip/hip_bf16.h>

// ---------------------------------------------------------------------------
// GAT layer, flash-attention formulation for MI455X (gfx1250, wave32, WMMA).
//   B=8, S=2048, F=128, H=4, D=32.  adj (134MB) is the roofline driver; heads
//   share adj tiles via LDS so HBM reads it once. All matmuls use
//   v_wmma_f32_16x16x32_f16 (D=32 == one WMMA K-step).
// ---------------------------------------------------------------------------

typedef __attribute__((ext_vector_type(16))) _Float16 v16h;
typedef __attribute__((ext_vector_type(8)))  float    v8f;

#define GAT_B 8
#define GAT_S 2048
#define GAT_F 128
#define GAT_H 4
#define GAT_D 32
#define NEG_INF_F (-9.0e15f)
#define INV_SQRT_D 0.17677669529663687f   // 1/sqrt(32)

// ---------------------------------------------------------------------------
// Kernel 1: Wh[b,h,s,d] = sum_f feat[b,s,f] * W[h,f,d], stored f16.
// One wave per (b,h, 16-row tile); two 16x16 C tiles cover D=32; K-loop over
// F=128 in 4 WMMA steps.
// ---------------------------------------------------------------------------
__global__ __launch_bounds__(32)
void gat_proj_kernel(const float* __restrict__ feat,
                     const float* __restrict__ W,
                     _Float16*    __restrict__ Wh) {
    const int id   = blockIdx.x;
    const int qt   = id % (GAT_S / 16);
    const int bh   = id / (GAT_S / 16);
    const int h    = bh % GAT_H;
    const int b    = bh / GAT_H;
    const int lane = threadIdx.x & 31;
    const int m    = lane & 15;       // A row / B,C col
    const int hi   = lane >> 4;       // half-wave select
    const int s0   = qt * 16;

    const float* frow = feat + ((size_t)b * GAT_S + s0 + m) * GAT_F;
    const float* Wm   = W + (size_t)h * GAT_F * GAT_D;

    v8f c0 = {};
    v8f c1 = {};

    for (int kc = 0; kc < GAT_F; kc += 32) {
        // A fragment: 16x32 f16, per-lane two contiguous runs of 8.
        v16h a;
#pragma unroll
        for (int i = 0; i < 8; ++i) {
            a[i]     = (_Float16)frow[kc +      8 * hi + i];
            a[8 + i] = (_Float16)frow[kc + 16 + 8 * hi + i];
        }
        // B fragments from W[h][f][d] (f = K, d = N): lane col n = m,
        // K = 8*hi + i (+8 for i>=8)  (strided gathers; W is 16KB/head -> cache resident)
        v16h b0, b1;
#pragma unroll
        for (int i = 0; i < 16; ++i) {
            const int kk = kc + 8 * hi + i + (i >= 8 ? 8 : 0);
            b0[i] = (_Float16)Wm[(size_t)kk * GAT_D + m];
            b1[i] = (_Float16)Wm[(size_t)kk * GAT_D + 16 + m];
        }
        c0 = __builtin_amdgcn_wmma_f32_16x16x32_f16(false, a, false, b0,
                                                    (short)0, c0, false, false);
        c1 = __builtin_amdgcn_wmma_f32_16x16x32_f16(false, a, false, b1,
                                                    (short)0, c1, false, false);
    }

    _Float16* out = Wh + (((size_t)b * GAT_H + h) * GAT_S + s0) * GAT_D;
#pragma unroll
    for (int r = 0; r < 8; ++r) {
        const int row = r + 8 * hi;                       // C layout
        out[(size_t)row * GAT_D + m]      = (_Float16)c0[r];
        out[(size_t)row * GAT_D + 16 + m] = (_Float16)c1[r];
    }
}

// ---------------------------------------------------------------------------
// Kernel 2: flash attention. Block = 128 threads = 4 waves, one per head,
// all sharing the same (b, 16-query-row tile). adj tile staged in LDS once
// per 32-key chunk for all 4 heads.
// ---------------------------------------------------------------------------
__global__ __launch_bounds__(128)
void gat_attn_kernel(const _Float16* __restrict__ Wh,
                     const int*      __restrict__ adj,
                     float*          __restrict__ out) {
    __shared__ int      adjTile[16 * 32];            // shared by all heads
    __shared__ _Float16 kvTile[GAT_H][32 * GAT_D];   // per-head K/V chunk
    __shared__ _Float16 pTile [GAT_H][16 * 32];      // per-head P (C->A bounce)

    const int blk  = blockIdx.x;
    const int qt   = blk % (GAT_S / 16);
    const int b    = blk / (GAT_S / 16);
    const int s0   = qt * 16;
    const int h    = threadIdx.x >> 5;    // wave == head
    const int lane = threadIdx.x & 31;
    const int m    = lane & 15;
    const int hi   = lane >> 4;

    const _Float16* WhH  = Wh + (((size_t)b * GAT_H + h) * GAT_S) * GAT_D;
    const int*      adjB = adj + ((size_t)b * GAT_S + s0) * GAT_S;

    // Q A-fragment (rows s0..s0+15, K = d = 0..31), loaded once.
    v16h aq;
    {
        const _Float16* qrow = WhH + (size_t)(s0 + m) * GAT_D;
#pragma unroll
        for (int i = 0; i < 8; ++i) {
            aq[i]     = qrow[     8 * hi + i];
            aq[8 + i] = qrow[16 + 8 * hi + i];
        }
    }

    v8f acc0 = {};
    v8f acc1 = {};
    float mrow[8], lrow[8];
#pragma unroll
    for (int r = 0; r < 8; ++r) { mrow[r] = -INFINITY; lrow[r] = 0.0f; }

    for (int kc = 0; kc < GAT_S; kc += 32) {
        __syncthreads();   // previous iteration done consuming adjTile
        // Cooperative adj tile load: 512 ints over 128 threads (coalesced).
        for (int i = (int)threadIdx.x; i < 16 * 32; i += 128) {
            const int rr = i >> 5, cc = i & 31;
            adjTile[i] = adjB[(size_t)rr * GAT_S + kc + cc];
        }
        // Stage this head's 32 K/V rows (f16, 64B/row) into LDS via b128s.
        {
            const uint4* src = (const uint4*)(WhH + (size_t)(kc + lane) * GAT_D);
            uint4*       dst = (uint4*)&kvTile[h][lane * GAT_D];
            dst[0] = src[0]; dst[1] = src[1]; dst[2] = src[2]; dst[3] = src[3];
        }
        // Prefetch next adj tile rows (global_prefetch_b8 on gfx1250).
        if (kc + 32 < GAT_S && threadIdx.x < 16)
            __builtin_prefetch(&adjB[(size_t)threadIdx.x * GAT_S + kc + 32], 0, 1);
        __syncthreads();

        // ----- scores: S = Q @ K^T, two 16x16 tiles (keys kc..+15, +16..+31)
        v16h bk0, bk1;
#pragma unroll
        for (int i = 0; i < 16; ++i) {            // B: n = key, K = d
            bk0[i] = kvTile[h][(size_t)m        * GAT_D + 8 * hi + i + (i >= 8 ? 8 : 0)];
            bk1[i] = kvTile[h][(size_t)(16 + m) * GAT_D + 8 * hi + i + (i >= 8 ? 8 : 0)];
        }
        v8f zero = {};
        v8f c0 = __builtin_amdgcn_wmma_f32_16x16x32_f16(false, aq, false, bk0,
                                                        (short)0, zero, false, false);
        v8f c1 = __builtin_amdgcn_wmma_f32_16x16x32_f16(false, aq, false, bk1,
                                                        (short)0, zero, false, false);

        // ----- mask + scale (finite -9e15, matching reference semantics)
        float sc0[8], sc1[8];
#pragma unroll
        for (int r = 0; r < 8; ++r) {
            const int row = r + 8 * hi;
            const int a0 = adjTile[row * 32 + m];
            const int a1 = adjTile[row * 32 + 16 + m];
            sc0[r] = (a0 > 0) ? c0[r] * INV_SQRT_D : NEG_INF_F;
            sc1[r] = (a1 > 0) ? c1[r] * INV_SQRT_D : NEG_INF_F;
        }

        // ----- online softmax; rows live in 16-lane halves -> xor 1/2/4/8
#pragma unroll
        for (int r = 0; r < 8; ++r) {
            float v = fmaxf(sc0[r], sc1[r]);
            v = fmaxf(v, __shfl_xor(v, 1, 32));
            v = fmaxf(v, __shfl_xor(v, 2, 32));
            v = fmaxf(v, __shfl_xor(v, 4, 32));
            v = fmaxf(v, __shfl_xor(v, 8, 32));
            const float mnew  = fmaxf(mrow[r], v);
            const float alpha = __expf(mrow[r] - mnew);   // exp(-inf)=0 first iter
            mrow[r] = mnew;
            const float p0 = __expf(sc0[r] - mnew);
            const float p1 = __expf(sc1[r] - mnew);
            float ls = p0 + p1;
            ls += __shfl_xor(ls, 1, 32);
            ls += __shfl_xor(ls, 2, 32);
            ls += __shfl_xor(ls, 4, 32);
            ls += __shfl_xor(ls, 8, 32);
            lrow[r] = lrow[r] * alpha + ls;
            acc0[r] *= alpha;
            acc1[r] *= alpha;
            const int row = r + 8 * hi;                   // C layout -> LDS
            pTile[h][row * 32 + m]      = (_Float16)p0;
            pTile[h][row * 32 + 16 + m] = (_Float16)p1;
        }

        // ----- O += P @ V: re-read P in A layout, V as B frags from LDS
        v16h ap;
#pragma unroll
        for (int i = 0; i < 8; ++i) {
            ap[i]     = pTile[h][m * 32 +      8 * hi + i];
            ap[8 + i] = pTile[h][m * 32 + 16 + 8 * hi + i];
        }
        v16h bv0, bv1;
#pragma unroll
        for (int i = 0; i < 16; ++i) {            // B: n = d, K = key
            const int key = 8 * hi + i + (i >= 8 ? 8 : 0);
            bv0[i] = kvTile[h][(size_t)key * GAT_D + m];
            bv1[i] = kvTile[h][(size_t)key * GAT_D + 16 + m];
        }
        acc0 = __builtin_amdgcn_wmma_f32_16x16x32_f16(false, ap, false, bv0,
                                                      (short)0, acc0, false, false);
        acc1 = __builtin_amdgcn_wmma_f32_16x16x32_f16(false, ap, false, bv1,
                                                      (short)0, acc1, false, false);
    }

    // Epilogue: out[b, s, h*D + d] = acc / l
    float* outB = out + ((size_t)b * GAT_S + s0) * (GAT_H * GAT_D) + h * GAT_D;
#pragma unroll
    for (int r = 0; r < 8; ++r) {
        const int row = r + 8 * hi;
        const float inv = 1.0f / lrow[r];
        outB[(size_t)row * (GAT_H * GAT_D) + m]      = acc0[r] * inv;
        outB[(size_t)row * (GAT_H * GAT_D) + 16 + m] = acc1[r] * inv;
    }
}

// ---------------------------------------------------------------------------
extern "C" void kernel_launch(void* const* d_in, const int* in_sizes, int n_in,
                              void* d_out, int out_size, void* d_ws, size_t ws_size,
                              hipStream_t stream) {
    (void)in_sizes; (void)n_in; (void)out_size; (void)ws_size;
    const float* feat = (const float*)d_in[0];   // [B,S,F] f32
    const int*   adj  = (const int*)d_in[1];     // [B,S,S] i32
    const float* W    = (const float*)d_in[2];   // [H,F,D] f32
    float*       out  = (float*)d_out;           // [B,S,H*D] f32
    _Float16*    Wh   = (_Float16*)d_ws;         // [B,H,S,D] f16 (4 MB)

    gat_proj_kernel<<<GAT_B * GAT_H * (GAT_S / 16), 32, 0, stream>>>(feat, W, Wh);
    gat_attn_kernel<<<GAT_B * (GAT_S / 16), 128, 0, stream>>>(Wh, adj, out);
}
